// Graphormer_59785944760736
// MI455X (gfx1250) — compile-verified, matrix operand order checked by reading, not run on hardware
//
#include <hip/hip_runtime.h>
#include <hip/hip_bf16.h>

typedef __attribute__((ext_vector_type(16))) _Float16 v16h;
typedef __attribute__((ext_vector_type(8)))  _Float16 v8h;
typedef __attribute__((ext_vector_type(8)))  float    v8f;
typedef __attribute__((ext_vector_type(4)))  unsigned int v4u;
typedef __attribute__((ext_vector_type(8)))  int      v8i;
typedef __attribute__((ext_vector_type(4)))  int      v4i;

constexpr int Dm   = 1024;   // model dim
constexpr int Hn   = 16;     // heads
constexpr int HDim = 64;     // head dim
constexpr int Sn   = 512;    // sequence (nodes)
constexpr int Bn   = 8;      // batch
constexpr int Lcnt = 12;     // layers
constexpr int Rows = Bn * Sn;          // 4096 activation rows
constexpr float QSCALE = 0.125f;       // 64^-0.5

// ---------------------------------------------------------------------------
// WMMA helpers (CDNA5 wave32, 16x16x32 f16 -> f32)
// ---------------------------------------------------------------------------
__device__ inline v8f wmma_f32_f16(v16h a, v16h b, v8f c) {
  // (neg_a, A, neg_b, B, c_mod, C, reuse_a, reuse_b)
  return __builtin_amdgcn_wmma_f32_16x16x32_f16(false, a, false, b, (short)0, c,
                                                false, false);
}

// 16x32 f16 fragment from a row-major matrix (leading dim `ld`, column k0).
// ISA layout: row M = lane&15; lane-group kg selects K 0-7/16-23 vs 8-15/24-31.
__device__ inline v16h load_frag16(const _Float16* base, int ld, int k0) {
  int lane = threadIdx.x & 31;
  int row  = lane & 15;
  int kg   = lane >> 4;
  const _Float16* p = base + (size_t)row * ld + k0 + kg * 8;
  v8h lo = *(const v8h*)p;
  v8h hi = *(const v8h*)(p + 16);
  return __builtin_shufflevector(lo, hi, 0,1,2,3,4,5,6,7,8,9,10,11,12,13,14,15);
}

// ---------------------------------------------------------------------------
// Tensor Data Mover: async 2D tile (tile_w_units x tile_rows, 8B units) from
// global row-major (row stride = stride_units) into LDS at lds_addr.
// D# layout per CDNA5 ISA ch.8: group0 = {count, lds_addr, global_addr, type},
// group1 = {data_size, tensor_dim0/1, tile_dim0/1/2, tensor_dim0_stride}.
// This toolchain's builtin takes 6 args (extra int32x8 group, then cpol).
// ---------------------------------------------------------------------------
__device__ inline void tdm_load_tile(unsigned int lds_addr, const _Float16* gsrc,
                                     unsigned int tile_w_units,
                                     unsigned int tile_rows,
                                     unsigned int stride_units) {
  unsigned long long ga = (unsigned long long)(uintptr_t)gsrc;
  v4u g0;
  g0.x = 1u;                                   // count=1, user-mode
  g0.y = lds_addr;                             // LDS byte address
  g0.z = (unsigned int)ga;                     // global_addr[31:0]
  g0.w = ((unsigned int)(ga >> 32) & 0x1FFFFFFu) | (2u << 30);  // addr[56:32]|type=2
  v8i g1;
  g1[0] = (int)(3u << 16);                     // data_size=8B, no multicast/pad
  g1[1] = (int)((stride_units & 0xFFFFu) << 16);               // tensor_dim0 lo16
  g1[2] = (int)(((stride_units >> 16) & 0xFFFFu) |
                ((tile_rows & 0xFFFFu) << 16));                // tdim0 hi | tdim1 lo
  g1[3] = (int)(((tile_rows >> 16) & 0xFFFFu) |
                ((tile_w_units & 0xFFFFu) << 16));             // tdim1 hi | tile_dim0
  g1[4] = (int)(tile_rows & 0xFFFFu);                          // tile_dim1 (tile_dim2=0)
  g1[5] = (int)stride_units;                                   // tensor_dim0_stride lo
  g1[6] = 0;                                                   // stride hi | stride1 lo
  g1[7] = 0;
  v4i g2 = {};
  v4i g3 = {};
  v8i g4 = {};                                 // unused trailing descriptor group
  __builtin_amdgcn_tensor_load_to_lds(g0, g1, g2, g3, g4, 0);
}

// ---------------------------------------------------------------------------
// Weight convert + transpose: W fp32 [K,N] (in,out) -> Wt f16 [N,K]
// ---------------------------------------------------------------------------
__global__ __launch_bounds__(256)
void transpose_cvt(const float* __restrict__ W, _Float16* __restrict__ Wt,
                   int K, int N) {
  __shared__ float tile[32][33];
  int tx = threadIdx.x & 31;
  int ty = threadIdx.x >> 5;
  int k0 = blockIdx.y * 32;
  int n0 = blockIdx.x * 32;
#pragma unroll
  for (int i = 0; i < 4; ++i) {
    int kk = ty + i * 8;
    tile[kk][tx] = W[(size_t)(k0 + kk) * N + n0 + tx];
  }
  __syncthreads();
#pragma unroll
  for (int i = 0; i < 4; ++i) {
    int nn = ty + i * 8;
    Wt[(size_t)(n0 + nn) * K + k0 + tx] = (_Float16)tile[tx][nn];
  }
}

// ---------------------------------------------------------------------------
// Block reduction (256 threads, wave32)
// ---------------------------------------------------------------------------
__device__ inline float block_sum256(float v) {
  __shared__ float red[8];
  int lane = threadIdx.x & 31;
  int w    = threadIdx.x >> 5;
#pragma unroll
  for (int o = 16; o > 0; o >>= 1) v += __shfl_xor(v, o, 32);
  __syncthreads();
  if (lane == 0) red[w] = v;
  __syncthreads();
  float tot = red[0];
#pragma unroll
  for (int i = 1; i < 8; ++i) tot += red[i];
  return tot;
}

// ---------------------------------------------------------------------------
// LayerNorm over D=1024, fp32 in, OutT out
// ---------------------------------------------------------------------------
template <typename OutT>
__global__ __launch_bounds__(256)
void ln_kernel(const float* __restrict__ x, const float* __restrict__ g,
               const float* __restrict__ be, OutT* __restrict__ y) {
  int row = blockIdx.x;
  const float* xr = x + (size_t)row * Dm;
  float v[4];
  float s = 0.f;
#pragma unroll
  for (int i = 0; i < 4; ++i) {
    v[i] = xr[threadIdx.x + 256 * i];
    s += v[i];
  }
  float mean = block_sum256(s) * (1.f / Dm);
  float var = 0.f;
#pragma unroll
  for (int i = 0; i < 4; ++i) {
    float d = v[i] - mean;
    var += d * d;
  }
  var = block_sum256(var) * (1.f / Dm);
  float inv = rsqrtf(var + 1e-5f);
#pragma unroll
  for (int i = 0; i < 4; ++i) {
    int c = threadIdx.x + 256 * i;
    y[(size_t)row * Dm + c] = (OutT)((v[i] - mean) * inv * g[c] + be[c]);
  }
}

// ---------------------------------------------------------------------------
// WMMA GEMM with TDM double-buffered LDS staging.
// C[M,N] = A[M,K](f16) * Wt[N,K]^T(f16) + bias, fused epilogues.
// Block = 8 waves -> 128 (M) x 64 (N) tile; K consumed in steps of 64.
// Wave 0 drives the Tensor Data Mover; TENSORcnt orders the pipeline
// (issue step s+1, s_wait_tensorcnt(2) retires step s's pair in order).
// MODE 0: Q-proj  -> f16 q[b,h,s,hd] * QSCALE
// MODE 1: K-proj  -> f16 k[b,h,s,hd]
// MODE 2: V-proj  -> f16 vT[b,h,hd,s]
// MODE 3: resid   -> f32 out = resid + val   (O-proj / FFN2)
// MODE 4: relu    -> f16 out = max(val,0)    (FFN1)
// ---------------------------------------------------------------------------
template <int MODE>
__global__ __launch_bounds__(256)
void gemm_wmma(const _Float16* __restrict__ A, const _Float16* __restrict__ Wt,
               const float* __restrict__ bias, void* __restrict__ out,
               const float* __restrict__ resid, int M, int N, int K) {
  __shared__ alignas(16) _Float16 ldsA[2][128 * 64];
  __shared__ alignas(16) _Float16 ldsB[2][64 * 64];
  int wave = threadIdx.x >> 5;
  int mblk = blockIdx.y * 128;
  int n0   = blockIdx.x * 64;
  const unsigned int strideU = (unsigned int)(K >> 2);  // K halfs * 2B / 8B units

  if (wave == 0) {
    tdm_load_tile((unsigned int)(uintptr_t)&ldsA[0][0], A + (size_t)mblk * K,
                  16, 128, strideU);
    tdm_load_tile((unsigned int)(uintptr_t)&ldsB[0][0], Wt + (size_t)n0 * K,
                  16, 64, strideU);
  }

  v8f acc[4] = {};
  const int steps = K >> 6;
  for (int s = 0; s < steps; ++s) {
    int cur = s & 1;
    if (wave == 0) {
      if (s + 1 < steps) {
        int nxt = cur ^ 1;
        int kk = (s + 1) << 6;
        tdm_load_tile((unsigned int)(uintptr_t)&ldsA[nxt][0],
                      A + (size_t)mblk * K + kk, 16, 128, strideU);
        tdm_load_tile((unsigned int)(uintptr_t)&ldsB[nxt][0],
                      Wt + (size_t)n0 * K + kk, 16, 64, strideU);
        __builtin_amdgcn_s_wait_tensorcnt(2);   // oldest pair (step s) retired
      } else {
        __builtin_amdgcn_s_wait_tensorcnt(0);
      }
    }
    __syncthreads();                            // LDS tiles visible to all waves
    const _Float16* pA = &ldsA[cur][wave * 16 * 64];
#pragma unroll
    for (int ks = 0; ks < 64; ks += 32) {
      v16h a = load_frag16(pA, 64, ks);
#pragma unroll
      for (int t = 0; t < 4; ++t) {
        v16h b = load_frag16(&ldsB[cur][t * 16 * 64], 64, ks);
        acc[t] = wmma_f32_f16(a, b, acc[t]);
      }
    }
    __syncthreads();                            // reads done before buffer reuse
  }

  int lane = threadIdx.x & 31;
  int lr = lane & 15;
  int kg = lane >> 4;
#pragma unroll
  for (int t = 0; t < 4; ++t) {
    int n = n0 + 16 * t + lr;
    float bv = bias[n];
#pragma unroll
    for (int r = 0; r < 8; ++r) {
      int m = mblk + wave * 16 + r + 8 * kg;
      float val = acc[t][r] + bv;
      if (MODE == 0 || MODE == 1 || MODE == 2) {
        int bidx = m >> 9;        // / Sn
        int s_   = m & 511;
        int hh   = n >> 6;        // / HDim
        int hd   = n & 63;
        _Float16* o = (_Float16*)out;
        if (MODE == 0)
          o[(((size_t)bidx * Hn + hh) * Sn + s_) * HDim + hd] =
              (_Float16)(val * QSCALE);
        else if (MODE == 1)
          o[(((size_t)bidx * Hn + hh) * Sn + s_) * HDim + hd] = (_Float16)val;
        else
          o[(((size_t)bidx * Hn + hh) * HDim + hd) * Sn + s_] = (_Float16)val;
      } else if (MODE == 3) {
        float* o = (float*)out;
        size_t idx = (size_t)m * N + n;
        o[idx] = resid[idx] + val;
      } else {  // MODE 4
        _Float16* o = (_Float16*)out;
        o[(size_t)m * N + n] = (_Float16)fmaxf(val, 0.f);
      }
    }
  }
}

// ---------------------------------------------------------------------------
// Flash-style attention. One wave per (b, h, 16 query rows).
// scores = q*k^T (+bias), online softmax, o = P*V with P relayout via LDS.
// ---------------------------------------------------------------------------
__global__ __launch_bounds__(32)
void attn_kernel(const _Float16* __restrict__ q, const _Float16* __restrict__ k,
                 const _Float16* __restrict__ vT, const float* __restrict__ bias,
                 _Float16* __restrict__ o) {
  __shared__ alignas(16) _Float16 ptile[16 * 32];
  int b  = blockIdx.z;
  int h  = blockIdx.y;
  int q0 = blockIdx.x * 16;
  int lane = threadIdx.x;
  int lr = lane & 15;
  int kg = lane >> 4;

  const _Float16* qbh = q  + (((size_t)b * Hn + h) * Sn) * HDim;
  const _Float16* kbh = k  + (((size_t)b * Hn + h) * Sn) * HDim;
  const _Float16* vbh = vT + (((size_t)b * Hn + h) * HDim) * Sn;
  const float* biash  = bias + ((size_t)h * Sn + q0) * Sn;

  v16h aq0 = load_frag16(qbh + (size_t)q0 * HDim, HDim, 0);
  v16h aq1 = load_frag16(qbh + (size_t)q0 * HDim, HDim, 32);

  v8f oacc[4] = {};
  float rmax[8], rsum[8];
#pragma unroll
  for (int r = 0; r < 8; ++r) { rmax[r] = -1e30f; rsum[r] = 0.f; }

  for (int kb = 0; kb < Sn / 32; ++kb) {
    int kbase = kb * 32;
    v8f c0 = {}, c1 = {};
    {
      v16h bk = load_frag16(kbh + (size_t)kbase * HDim, HDim, 0);
      c0 = wmma_f32_f16(aq0, bk, c0);
      bk = load_frag16(kbh + (size_t)kbase * HDim, HDim, 32);
      c0 = wmma_f32_f16(aq1, bk, c0);
      bk = load_frag16(kbh + (size_t)(kbase + 16) * HDim, HDim, 0);
      c1 = wmma_f32_f16(aq0, bk, c1);
      bk = load_frag16(kbh + (size_t)(kbase + 16) * HDim, HDim, 32);
      c1 = wmma_f32_f16(aq1, bk, c1);
    }
    float s0[8], s1[8], alpha[8];
#pragma unroll
    for (int r = 0; r < 8; ++r) {
      int m = r + 8 * kg;
      s0[r] = c0[r] + biash[(size_t)m * Sn + kbase + lr];
      s1[r] = c1[r] + biash[(size_t)m * Sn + kbase + 16 + lr];
    }
#pragma unroll
    for (int r = 0; r < 8; ++r) {
      float mx = fmaxf(s0[r], s1[r]);
#pragma unroll
      for (int off = 8; off > 0; off >>= 1)
        mx = fmaxf(mx, __shfl_xor(mx, off, 32));   // within 16-lane row group
      mx = fmaxf(mx, rmax[r]);
      alpha[r] = __expf(rmax[r] - mx);
      rmax[r] = mx;
      float p0 = __expf(s0[r] - mx);
      float p1 = __expf(s1[r] - mx);
      s0[r] = p0; s1[r] = p1;
      float ps = p0 + p1;
#pragma unroll
      for (int off = 8; off > 0; off >>= 1) ps += __shfl_xor(ps, off, 32);
      rsum[r] = rsum[r] * alpha[r] + ps;
    }
    // C-fragment -> A-fragment relayout through LDS
    __syncthreads();
#pragma unroll
    for (int r = 0; r < 8; ++r) {
      int m = r + 8 * kg;
      ptile[m * 32 + lr]      = (_Float16)s0[r];
      ptile[m * 32 + 16 + lr] = (_Float16)s1[r];
    }
    __syncthreads();
    v16h pa;
    {
      const _Float16* p = &ptile[lr * 32 + kg * 8];
      v8h lo = *(const v8h*)p;
      v8h hi = *(const v8h*)(p + 16);
      pa = __builtin_shufflevector(lo, hi, 0,1,2,3,4,5,6,7,8,9,10,11,12,13,14,15);
    }
#pragma unroll
    for (int t = 0; t < 4; ++t) {
#pragma unroll
      for (int r = 0; r < 8; ++r) oacc[t][r] *= alpha[r];
      v16h bv = load_frag16(vbh + (size_t)(16 * t) * Sn, Sn, kbase);
      oacc[t] = wmma_f32_f16(pa, bv, oacc[t]);
    }
  }
#pragma unroll
  for (int t = 0; t < 4; ++t) {
    int hd = 16 * t + lr;
#pragma unroll
    for (int r = 0; r < 8; ++r) {
      int m = r + 8 * kg;
      float val = oacc[t][r] / rsum[r];
      o[((size_t)b * Sn + q0 + m) * Dm + h * HDim + hd] = (_Float16)val;
    }
  }
}

// ---------------------------------------------------------------------------
// Head: mean pool, fc1+relu, LN (ln_kernel<float>), fc2
// ---------------------------------------------------------------------------
__global__ __launch_bounds__(256)
void pool_kernel(const float* __restrict__ x, float* __restrict__ z) {
  int d = blockIdx.x * 256 + threadIdx.x;
  int b = blockIdx.y;
  float s = 0.f;
  for (int i = 0; i < Sn; ++i) s += x[((size_t)b * Sn + i) * Dm + d];
  z[(size_t)b * Dm + d] = s * (1.f / Sn);
}

__global__ __launch_bounds__(256)
void fc_relu_kernel(const float* __restrict__ z, const float* __restrict__ W,
                    const float* __restrict__ bias, float* __restrict__ out) {
  int n = blockIdx.x * 256 + threadIdx.x;
  int b = blockIdx.y;
  float s = bias[n];
  for (int kk = 0; kk < Dm; ++kk) s += z[(size_t)b * Dm + kk] * W[(size_t)kk * Dm + n];
  out[(size_t)b * Dm + n] = fmaxf(s, 0.f);
}

__global__ __launch_bounds__(128)
void fc_out_kernel(const float* __restrict__ z, const float* __restrict__ W,
                   const float* __restrict__ bias, float* __restrict__ out) {
  int t = threadIdx.x;
  if (t >= Bn * 10) return;
  int b = t / 10, c = t % 10;
  float s = bias[c];
  for (int kk = 0; kk < Dm; ++kk) s += z[(size_t)b * Dm + kk] * W[(size_t)kk * 10 + c];
  out[t] = s;
}

// ---------------------------------------------------------------------------
// Host orchestration
// ---------------------------------------------------------------------------
extern "C" void kernel_launch(void* const* d_in, const int* in_sizes, int n_in,
                              void* d_out, int out_size, void* d_ws, size_t ws_size,
                              hipStream_t stream) {
  const float* x    = (const float*)d_in[0];
  const float* bias = (const float*)d_in[1];
  const float* Wq = (const float*)d_in[2];  const float* bq = (const float*)d_in[3];
  const float* Wk = (const float*)d_in[4];  const float* bk = (const float*)d_in[5];
  const float* Wv = (const float*)d_in[6];  const float* bv = (const float*)d_in[7];
  const float* Wo = (const float*)d_in[8];  const float* bo = (const float*)d_in[9];
  const float* W1 = (const float*)d_in[10]; const float* b1 = (const float*)d_in[11];
  const float* W2 = (const float*)d_in[12]; const float* b2 = (const float*)d_in[13];
  const float* g1 = (const float*)d_in[14]; const float* be1 = (const float*)d_in[15];
  const float* g2 = (const float*)d_in[16]; const float* be2 = (const float*)d_in[17];
  const float* Wf1 = (const float*)d_in[18]; const float* bf1 = (const float*)d_in[19];
  const float* gf  = (const float*)d_in[20]; const float* bf  = (const float*)d_in[21];
  const float* Wf2 = (const float*)d_in[22]; const float* bf2 = (const float*)d_in[23];
  (void)in_sizes; (void)n_in; (void)out_size; (void)ws_size;

  char* ws = (char*)d_ws;
  size_t off = 0;
  auto carve = [&](size_t bytes) {
    char* p = ws + off;
    off += (bytes + 255) & ~(size_t)255;
    return p;
  };
  float*    xw    = (float*)   carve((size_t)Rows * Dm * 4);       // residual stream
  _Float16* y     = (_Float16*)carve((size_t)Rows * Dm * 2);       // LN output
  _Float16* qh    = (_Float16*)carve((size_t)Rows * Dm * 2);       // [b,h,s,hd]
  _Float16* kh    = (_Float16*)carve((size_t)Rows * Dm * 2);       // [b,h,s,hd]
  _Float16* vth   = (_Float16*)carve((size_t)Rows * Dm * 2);       // [b,h,hd,s]
  _Float16* attnh = (_Float16*)carve((size_t)Rows * Dm * 2);       // [b,s,D]
  _Float16* ffnh  = (_Float16*)carve((size_t)Rows * 4 * Dm * 2);   // [4096,4096]
  _Float16* wt    = (_Float16*)carve((size_t)4 * Dm * Dm * 2);     // transposed weight
  float*    z0    = (float*)   carve((size_t)Bn * Dm * 4);
  float*    z1    = (float*)   carve((size_t)Bn * Dm * 4);
  float*    z2    = (float*)   carve((size_t)Bn * Dm * 4);

  (void)hipMemcpyAsync(xw, x, (size_t)Rows * Dm * 4, hipMemcpyDeviceToDevice, stream);

  dim3 blk(256);
  dim3 gT_DD(Dm / 32, Dm / 32);        // transpose [D,D]
  dim3 gT_D4D(4 * Dm / 32, Dm / 32);   // transpose [D,4D]
  dim3 gT_4DD(Dm / 32, 4 * Dm / 32);   // transpose [4D,D]
  dim3 gG_D(Dm / 64, Rows / 128);      // GEMM N=1024
  dim3 gG_4D(4 * Dm / 64, Rows / 128); // GEMM N=4096
  dim3 gAttn(Sn / 16, Hn, Bn);

  for (int l = 0; l < Lcnt; ++l) {
    const float* wq = Wq + (size_t)l * Dm * Dm;
    const float* wk = Wk + (size_t)l * Dm * Dm;
    const float* wv = Wv + (size_t)l * Dm * Dm;
    const float* wo = Wo + (size_t)l * Dm * Dm;
    const float* w1 = W1 + (size_t)l * Dm * 4 * Dm;
    const float* w2 = W2 + (size_t)l * 4 * Dm * Dm;

    // --- attention block ---
    ln_kernel<_Float16><<<Rows, blk, 0, stream>>>(xw, g1 + l * Dm, be1 + l * Dm, y);

    transpose_cvt<<<gT_DD, blk, 0, stream>>>(wq, wt, Dm, Dm);
    gemm_wmma<0><<<gG_D, blk, 0, stream>>>(y, wt, bq + l * Dm, qh, nullptr, Rows, Dm, Dm);

    transpose_cvt<<<gT_DD, blk, 0, stream>>>(wk, wt, Dm, Dm);
    gemm_wmma<1><<<gG_D, blk, 0, stream>>>(y, wt, bk + l * Dm, kh, nullptr, Rows, Dm, Dm);

    transpose_cvt<<<gT_DD, blk, 0, stream>>>(wv, wt, Dm, Dm);
    gemm_wmma<2><<<gG_D, blk, 0, stream>>>(y, wt, bv + l * Dm, vth, nullptr, Rows, Dm, Dm);

    attn_kernel<<<gAttn, dim3(32), 0, stream>>>(qh, kh, vth, bias, attnh);

    transpose_cvt<<<gT_DD, blk, 0, stream>>>(wo, wt, Dm, Dm);
    gemm_wmma<3><<<gG_D, blk, 0, stream>>>(attnh, wt, bo + l * Dm, xw, xw, Rows, Dm, Dm);

    // --- FFN block ---
    ln_kernel<_Float16><<<Rows, blk, 0, stream>>>(xw, g2 + l * Dm, be2 + l * Dm, y);

    transpose_cvt<<<gT_D4D, blk, 0, stream>>>(w1, wt, Dm, 4 * Dm);
    gemm_wmma<4><<<gG_4D, blk, 0, stream>>>(y, wt, b1 + (size_t)l * 4 * Dm, ffnh, nullptr,
                                            Rows, 4 * Dm, Dm);

    transpose_cvt<<<gT_4DD, blk, 0, stream>>>(w2, wt, 4 * Dm, Dm);
    gemm_wmma<3><<<gG_D, blk, 0, stream>>>(ffnh, wt, b2 + (size_t)l * Dm, xw, xw,
                                           Rows, Dm, 4 * Dm);
  }

  // --- head ---
  pool_kernel<<<dim3(Dm / 256, Bn), blk, 0, stream>>>(xw, z0);
  fc_relu_kernel<<<dim3(Dm / 256, Bn), blk, 0, stream>>>(z0, Wf1, bf1, z1);
  ln_kernel<float><<<Bn, blk, 0, stream>>>(z1, gf, bf, z2);
  fc_out_kernel<<<1, 128, 0, stream>>>(z2, Wf2, bf2, (float*)d_out);
}
